// SparseMultiHeadAttention_14113262534999
// MI455X (gfx1250) — matrix operand order repaired and checked by reading, not compiled, not run on hardware
//
#include <hip/hip_runtime.h>
#include <hip/hip_bf16.h>
#include <math.h>
#include <stdint.h>

typedef _Float16 half_t;
typedef __attribute__((ext_vector_type(16))) _Float16 v16h;
typedef __attribute__((ext_vector_type(8)))  float    v8f;

// Problem constants (match reference)
#define NPTS 55296
#define CCH  512
#define HH   8
#define DD   64
#define NWIN_AX 13
#define NWIN (13*13*13)   // 2197
#define CAPW 64

// ---------------------------------------------------------------------------
// WMMA fragment loaders per CDNA5 ISA 7.12.2 (wave32):
//  A 16x32 f16: lane L<16 holds row M=L, K = 0..7 and 16..23 (VGPR0..3 / 4..7)
//               lane L>=16 holds row M=L-16, K = 8..15 and 24..31
//  B 32x16 f16 (given transposed storage Bt[n][k]): lane L<16 holds col N=L,
//               K = 0..7 and 16..23; lane L>=16 holds col N=L-16, K = 8..15 and 24..31.
//  C/D 16x16 f32: lane L, vgpr r -> (M = r + 8*(L>=16), N = L&15)
// ---------------------------------------------------------------------------
__device__ static inline v16h load_frag_a(const half_t* __restrict__ A, int lda) {
    int lane = threadIdx.x & 31;
    int r  = lane & 15;
    int kb = (lane >> 4) * 8;
    const half_t* p = A + (size_t)r * lda;
    v16h a;
#pragma unroll
    for (int j = 0; j < 8; ++j) a[j]     = p[kb + j];
#pragma unroll
    for (int j = 0; j < 8; ++j) a[8 + j] = p[16 + kb + j];
    return a;
}

__device__ static inline v16h load_frag_b(const half_t* __restrict__ Bt, int ldb) {
    int lane = threadIdx.x & 31;
    int n   = lane & 15;
    int kb  = (lane >> 4) * 8;
    const half_t* p = Bt + (size_t)n * ldb + kb;
    v16h b;
#pragma unroll
    for (int j = 0; j < 8; ++j) b[j] = p[j];
#pragma unroll
    for (int j = 0; j < 8; ++j) b[8 + j] = p[16 + j];
    return b;
}

#define WMMA_F16(a, b, c) \
    __builtin_amdgcn_wmma_f32_16x16x32_f16(false, (a), false, (b), (short)0, (c), false, false)

// ---------------------------------------------------------------------------
// 0) init workspace state
// ---------------------------------------------------------------------------
__global__ void init_kernel(int* __restrict__ counts, int* __restrict__ pslot) {
    int i = blockIdx.x * 256 + threadIdx.x;
    if (i < NWIN * CAPW) pslot[i] = -1;
    if (i < NWIN)        counts[i] = 0;
}

// ---------------------------------------------------------------------------
// 1) window binning via atomics (order inside a window is irrelevant)
// ---------------------------------------------------------------------------
__global__ void bin_kernel(const int* __restrict__ coords,
                           int* __restrict__ counts, int* __restrict__ pslot) {
    int i = blockIdx.x * 256 + threadIdx.x;
    if (i >= NPTS) return;
    int w0 = (coords[3*i+0] + 2) >> 2;
    int w1 = (coords[3*i+1] + 2) >> 2;
    int w2 = (coords[3*i+2] + 2) >> 2;
    int wid = (w0 * NWIN_AX + w1) * NWIN_AX + w2;
    int pos = atomicAdd(&counts[wid], 1);
    pslot[wid * CAPW + pos] = i;
}

// ---------------------------------------------------------------------------
// 2) f32 -> f16 conversion
// ---------------------------------------------------------------------------
__global__ void cvt_kernel(const float* __restrict__ src, half_t* __restrict__ dst, int n) {
    int i = blockIdx.x * 256 + threadIdx.x;
    if (i < n) dst[i] = (half_t)src[i];
}

// ---------------------------------------------------------------------------
// 3) WMMA GEMM: C[m][n] = sum_k A[m][k] * Bt[n][k] + bias[n]
//    block = 8 waves; wave w owns a 16x64 strip (4 accumulators), so each
//    A fragment feeds 4 WMMAs. Block covers 16 x 512 of C.
// ---------------------------------------------------------------------------
__global__ __launch_bounds__(256) void gemm16_kernel(
    const half_t* __restrict__ A, const half_t* __restrict__ Bt,
    const float* __restrict__ bias,
    half_t* __restrict__ Ch, float* __restrict__ Cf,
    int K, int ldc) {
    int wave = threadIdx.x >> 5;
    int m0 = blockIdx.x * 16;
    int n0 = blockIdx.y * 512 + wave * 64;
    const half_t* Ap = A  + (size_t)m0 * K;
    const half_t* Bp = Bt + (size_t)n0 * K;
    v8f acc0 = {}, acc1 = {}, acc2 = {}, acc3 = {};
    for (int k0 = 0; k0 < K; k0 += 32) {
        __builtin_prefetch(Ap + k0 + 128, 0, 1);   // global_prefetch_b8
        __builtin_prefetch(Bp + k0 + 128, 0, 1);
        v16h a  = load_frag_a(Ap + k0, K);
        v16h b0 = load_frag_b(Bp + k0, K);
        v16h b1 = load_frag_b(Bp + (size_t)16 * K + k0, K);
        v16h b2 = load_frag_b(Bp + (size_t)32 * K + k0, K);
        v16h b3 = load_frag_b(Bp + (size_t)48 * K + k0, K);
        acc0 = WMMA_F16(a, b0, acc0);
        acc1 = WMMA_F16(a, b1, acc1);
        acc2 = WMMA_F16(a, b2, acc2);
        acc3 = WMMA_F16(a, b3, acc3);
    }
    int lane = threadIdx.x & 31;
    int n  = lane & 15;
    int mh = (lane >> 4) * 8;
    v8f accs[4] = {acc0, acc1, acc2, acc3};
#pragma unroll
    for (int nt = 0; nt < 4; ++nt) {
        int nc = n0 + nt * 16 + n;
        float bv = bias[nc];
#pragma unroll
        for (int r = 0; r < 8; ++r) {
            size_t idx = (size_t)(m0 + mh + r) * ldc + nc;
            float v = accs[nt][r] + bv;
            if (Ch) Ch[idx] = (half_t)v;
            else    Cf[idx] = v;
        }
    }
}

// ---------------------------------------------------------------------------
// 4) RMS norm of q and k rows (in place, f16 buffer, f32 math), wave per row
// ---------------------------------------------------------------------------
__global__ __launch_bounds__(256) void rmsnorm_kernel(half_t* __restrict__ qkvh,
                                                      const float* __restrict__ gq,
                                                      const float* __restrict__ gk) {
    int wave = threadIdx.x >> 5;
    int lane = threadIdx.x & 31;
    long rid = (long)blockIdx.x * 8 + wave;           // 0 .. 2*NPTS*HH-1
    int isK = rid >= (long)NPTS * HH;
    long r = isK ? rid - (long)NPTS * HH : rid;
    int p = (int)(r / HH), h = (int)(r % HH);
    half_t* row = qkvh + (size_t)p * (3 * CCH) + isK * CCH + h * DD;
    float v0 = (float)row[2 * lane];
    float v1 = (float)row[2 * lane + 1];
    float ss = v0 * v0 + v1 * v1;
#pragma unroll
    for (int off = 16; off; off >>= 1) ss += __shfl_xor(ss, off, 32);
    float nrm = fmaxf(sqrtf(ss), 1e-12f);
    const float* g = isK ? gk : gq;
    float s = 8.0f / nrm;                              // sqrt(D)=8
    row[2 * lane]     = (half_t)(v0 * s * g[h * DD + 2 * lane]);
    row[2 * lane + 1] = (half_t)(v1 * s * g[h * DD + 2 * lane + 1]);
}

// ---------------------------------------------------------------------------
// 5) windowed attention: one block per (window, head). 64x64 per head.
//    Q/K rows staged into LDS via GLOBAL_LOAD_ASYNC_TO_LDS_B128 (ASYNCcnt),
//    V staged transposed via regular ds stores.
//    S = Q K^T * 1/8, mask invalid, softmax, O = P V. All matmuls via WMMA.
// ---------------------------------------------------------------------------
__global__ __launch_bounds__(256) void attn_kernel(const half_t* __restrict__ qkvh,
                                                   const int* __restrict__ pslot,
                                                   half_t* __restrict__ aoh) {
    __shared__ half_t Qs[64][72];
    __shared__ half_t Ks[64][72];
    __shared__ half_t Vt[64][72];   // Vt[d][slot]
    __shared__ float  Sm[64][68];   // scores, then output
    __shared__ half_t Pm[64][72];   // softmax probs (f16)
    __shared__ int    validm[64];
    __shared__ int    pidm[64];

    int w = blockIdx.x >> 3;
    int h = blockIdx.x & 7;
    int tid = threadIdx.x;
    int wave = tid >> 5, lane = tid & 31;

    // ---- gather ----
    int s  = tid >> 2;
    int d0 = (tid & 3) * 16;
    int p  = pslot[w * CAPW + s];
    const half_t* qrow = qkvh + (size_t)(p < 0 ? 0 : p) * (3 * CCH) + h * DD;
    if (p >= 0) {
        // 32B per row chunk: two async b128 copies straight into LDS.
        // IOFFSET is applied to both the LDS and the global address.
        uint32_t q_lds = (uint32_t)(uintptr_t)&Qs[s][d0];
        uint32_t k_lds = (uint32_t)(uintptr_t)&Ks[s][d0];
        const half_t* qg = qrow + d0;
        const half_t* kg = qrow + CCH + d0;
        asm volatile("global_load_async_to_lds_b128 %0, %1, off\n\t"
                     "global_load_async_to_lds_b128 %0, %1, off offset:16"
                     :: "v"(q_lds), "v"(qg) : "memory");
        asm volatile("global_load_async_to_lds_b128 %0, %1, off\n\t"
                     "global_load_async_to_lds_b128 %0, %1, off offset:16"
                     :: "v"(k_lds), "v"(kg) : "memory");
#pragma unroll
        for (int j = 0; j < 16; ++j)
            Vt[d0 + j][s] = qrow[2 * CCH + d0 + j];
    } else {
#pragma unroll
        for (int j = 0; j < 16; ++j) {
            Qs[s][d0 + j] = (half_t)0.f;
            Ks[s][d0 + j] = (half_t)0.f;
            Vt[d0 + j][s] = (half_t)0.f;
        }
    }
    if ((tid & 3) == 0) { validm[s] = (p >= 0); pidm[s] = p; }
    asm volatile("s_wait_asynccnt 0x0" ::: "memory");
    __syncthreads();

    // ---- pass 1: S = Q K^T ----
    for (int t = wave; t < 16; t += 8) {
        int mt = t >> 2, nt = t & 3;
        v8f acc = {};
#pragma unroll
        for (int k0 = 0; k0 < 64; k0 += 32) {
            v16h a = load_frag_a(&Qs[mt * 16][k0], 72);
            v16h b = load_frag_b(&Ks[nt * 16][k0], 72);
            acc = WMMA_F16(a, b, acc);
        }
        int n = lane & 15, mh = (lane >> 4) * 8;
#pragma unroll
        for (int r = 0; r < 8; ++r) Sm[mt * 16 + mh + r][nt * 16 + n] = acc[r];
    }
    __syncthreads();

    // ---- softmax over k (masked), 64 threads each own one row ----
    if (tid < 64) {
        const float sc = 0.125f;                       // 1/sqrt(64)
        float mx = -3e38f;
        for (int j = 0; j < 64; ++j) {
            float sv = validm[j] ? Sm[tid][j] * sc : -1e30f;
            Sm[tid][j] = sv;
            mx = fmaxf(mx, sv);
        }
        float sum = 0.f;
        for (int j = 0; j < 64; ++j) {
            float e = __expf(Sm[tid][j] - mx);
            Sm[tid][j] = e;
            sum += e;
        }
        float inv = 1.f / fmaxf(sum, 1e-30f);
        for (int j = 0; j < 64; ++j) Pm[tid][j] = (half_t)(Sm[tid][j] * inv);
    }
    __syncthreads();

    // ---- pass 2: O = P V  (B operand = Vt[d][slot]) ----
    for (int t = wave; t < 16; t += 8) {
        int mt = t >> 2, nt = t & 3;
        v8f acc = {};
#pragma unroll
        for (int k0 = 0; k0 < 64; k0 += 32) {
            v16h a = load_frag_a(&Pm[mt * 16][k0], 72);
            v16h b = load_frag_b(&Vt[nt * 16][k0], 72);
            acc = WMMA_F16(a, b, acc);
        }
        int n = lane & 15, mh = (lane >> 4) * 8;
#pragma unroll
        for (int r = 0; r < 8; ++r) Sm[mt * 16 + mh + r][nt * 16 + n] = acc[r];
    }
    __syncthreads();

    // ---- scatter valid rows back ----
    if (validm[s]) {
        half_t* orow = aoh + (size_t)pidm[s] * CCH + h * DD;
#pragma unroll
        for (int j = 0; j < 16; ++j) orow[d0 + j] = (half_t)Sm[s][d0 + j];
    }
}

// ---------------------------------------------------------------------------
// host-side launch
// ---------------------------------------------------------------------------
extern "C" void kernel_launch(void* const* d_in, const int* in_sizes, int n_in,
                              void* d_out, int out_size, void* d_ws, size_t ws_size,
                              hipStream_t stream) {
    (void)in_sizes; (void)n_in; (void)out_size; (void)ws_size;
    const float* x      = (const float*)d_in[0];
    const int*   coords = (const int*)  d_in[1];
    const float* Wqkv   = (const float*)d_in[2];
    const float* bqkv   = (const float*)d_in[3];
    const float* gq     = (const float*)d_in[4];
    const float* gk     = (const float*)d_in[5];
    const float* Wo     = (const float*)d_in[6];
    const float* bo     = (const float*)d_in[7];
    float* out = (float*)d_out;

    char* base = (char*)d_ws;
    size_t off = 0;
    auto take = [&](size_t bytes) -> char* {
        char* r = base + off;
        off = (off + bytes + 255) & ~(size_t)255;
        return r;
    };
    int*    counts = (int*)   take((size_t)NWIN * 4);
    int*    pslot  = (int*)   take((size_t)NWIN * CAPW * 4);
    half_t* xh     = (half_t*)take((size_t)NPTS * CCH * 2);
    half_t* wqkvh  = (half_t*)take((size_t)3 * CCH * CCH * 2);
    half_t* woh    = (half_t*)take((size_t)CCH * CCH * 2);
    half_t* qkvh   = (half_t*)take((size_t)NPTS * 3 * CCH * 2);
    half_t* aoh    = (half_t*)take((size_t)NPTS * CCH * 2);

    init_kernel<<<(NWIN * CAPW + 255) / 256, 256, 0, stream>>>(counts, pslot);
    bin_kernel <<<(NPTS + 255) / 256, 256, 0, stream>>>(coords, counts, pslot);

    cvt_kernel<<<(NPTS * CCH + 255) / 256, 256, 0, stream>>>(x, xh, NPTS * CCH);
    cvt_kernel<<<(3 * CCH * CCH + 255) / 256, 256, 0, stream>>>(Wqkv, wqkvh, 3 * CCH * CCH);
    cvt_kernel<<<(CCH * CCH + 255) / 256, 256, 0, stream>>>(Wo, woh, CCH * CCH);

    // qkv = x @ Wqkv^T + bqkv  (f16 out); N = 1536 -> grid.y = 3
    gemm16_kernel<<<dim3(NPTS / 16, (3 * CCH) / 512), 256, 0, stream>>>(
        xh, wqkvh, bqkv, qkvh, nullptr, CCH, 3 * CCH);

    rmsnorm_kernel<<<(2 * NPTS * HH) / 8, 256, 0, stream>>>(qkvh, gq, gk);

    attn_kernel<<<NWIN * HH, 256, 0, stream>>>(qkvh, pslot, aoh);

    // out = h @ Wo^T + bo  (f32 out); N = 512 -> grid.y = 1
    gemm16_kernel<<<dim3(NPTS / 16, CCH / 512), 256, 0, stream>>>(
        aoh, woh, bo, nullptr, out, CCH, CCH);
}
